// SelfAttention_39599598469416
// MI455X (gfx1250) — matrix operand order repaired and checked by reading, not compiled, not run on hardware
//
#include <hip/hip_runtime.h>

// ---------------------------------------------------------------------------
// CDNA5 / gfx1250 implementation. All big GEMMs run on v_wmma_f32_16x16x32_bf16.
// GEMM waves compute 16x64 strips with explicit 2-deep software pipelining:
// k-step i+1's A/B fragments are loaded before k-step i's 4 WMMAs issue.
// ---------------------------------------------------------------------------

#define DEV __device__ __forceinline__

typedef unsigned short u16;
typedef unsigned int   u32;
typedef __attribute__((ext_vector_type(16))) __bf16 v16bf;
typedef __attribute__((ext_vector_type(8)))  float  v8f;
typedef __attribute__((ext_vector_type(8)))  u32    v8u;
typedef __attribute__((ext_vector_type(4)))  u32    v4u;

union Frag {           // one 16x32 bf16 A-fragment or 32x16 bf16 B-fragment
  v16bf bf;
  u16   u[16];
  v4u   q[2];
  v8u   o;
};

DEV u16 f2bf(float f) {                 // round-to-nearest-even f32 -> bf16
  u32 x = __float_as_uint(f);
  u32 r = x + 0x7FFFu + ((x >> 16) & 1u);
  return (u16)(r >> 16);
}
DEV float bf2f(u16 h) { return __uint_as_float(((u32)h) << 16); }

DEV v8f wmma_bf(const Frag& a, const Frag& b, v8f c) {
  return __builtin_amdgcn_wmma_f32_16x16x32_bf16(false, a.bf, false, b.bf,
                                                 (short)0, c, false, false);
}

// ------------------------- elementwise prep kernels -------------------------

__global__ void cvt_bf16_kernel(const float* __restrict__ in,
                                u16* __restrict__ out, int n) {
  int i = blockIdx.x * 256 + threadIdx.x;
  if (i < n) out[i] = f2bf(in[i]);
}

// weight (768x768, row=k, col=n) -> bf16 transposed WT[n*768+k]
__global__ void cvt_T768_kernel(const float* __restrict__ in,
                                u16* __restrict__ out) {
  int i = blockIdx.x * 256 + threadIdx.x;
  if (i < 768 * 768) {
    int n = i / 768, k = i % 768;
    out[i] = f2bf(in[k * 768 + n]);
  }
}

// cand[(b,t,i)] rows: i==0 -> x[b,t+1], else neighbour_rep[b,t,i-1]
__global__ void gather_cand_kernel(const float* __restrict__ x,
                                   const float* __restrict__ nrep,
                                   u16* __restrict__ out) {
  size_t i = (size_t)blockIdx.x * 256 + threadIdx.x;
  if (i >= (size_t)20440 * 768) return;
  int r = (int)(i / 768), c = (int)(i % 768);
  int b = r / 2555, rem = r % 2555;
  int tk = rem / 5, ii = rem % 5;
  const float* src = (ii == 0)
      ? x + ((size_t)b * 512 + tk + 1) * 768
      : nrep + (((size_t)(b * 511 + tk)) * 4 + (ii - 1)) * 768;
  out[i] = f2bf(src[c]);
}

// Vb (B*512 x 768 bf16) -> Vt[b*768+c][key]  (B,768,512) bf16
__global__ void transpose_vt_kernel(const u16* __restrict__ Vb,
                                    u16* __restrict__ Vt) {
  size_t i = (size_t)blockIdx.x * 256 + threadIdx.x;
  if (i >= (size_t)8 * 768 * 512) return;
  int b = (int)(i / (768 * 512));
  int rem = (int)(i % (768 * 512));
  int c = rem / 512, key = rem % 512;
  Vt[i] = Vb[((size_t)(b * 512 + key)) * 768 + c];
}

// xd = A(bf16 Mx768) @ Wd(f32 768x8) + bd
__global__ void gemv_rank8_kernel(const u16* __restrict__ A, int M,
                                  const float* __restrict__ Wd,
                                  const float* __restrict__ bd,
                                  float* __restrict__ out) {
  int g = blockIdx.x * 256 + threadIdx.x;
  if (g >= M * 8) return;
  int row = g >> 3, rr = g & 7;
  const u16* a = A + (size_t)row * 768;
  float acc = 0.f;
  for (int k = 0; k < 768; ++k) acc += bf2f(a[k]) * Wd[k * 8 + rr];
  out[g] = acc + bd[rr];
}

// out = xd(f32 Mx8) @ Wu(f32 8x768) + bu  -> bf16
__global__ void lowrank_up_kernel(const float* __restrict__ xd, int M,
                                  const float* __restrict__ Wu,
                                  const float* __restrict__ bu,
                                  u16* __restrict__ out) {
  size_t i = (size_t)blockIdx.x * 256 + threadIdx.x;
  if (i >= (size_t)M * 768) return;
  int row = (int)(i / 768), n = (int)(i % 768);
  float acc = bu[n];
  const float* xr = xd + (size_t)row * 8;
  #pragma unroll
  for (int r = 0; r < 8; ++r) acc += xr[r] * Wu[r * 768 + n];
  out[i] = f2bf(acc);
}

// ------------------------------ WMMA GEMM -----------------------------------
// out(MxN=768) = A(Mx768 bf16) @ W + bias ; WT[n*768+k] = W[k][n]
// Each wave computes a 16x64 strip; 2-deep software pipeline over k.
__global__ void __launch_bounds__(256)
gemm_bf16_768(const u16* __restrict__ A, int M,
              const u16* __restrict__ WT,
              const float* __restrict__ bias,
              float* __restrict__ outF, u16* __restrict__ outB) {
  const int lane = threadIdx.x & 31, wave = threadIdx.x >> 5;
  const int lo = lane & 15, hi = lane >> 4;
  const int Mtiles = (M + 15) >> 4;
  const int tile = blockIdx.x * 8 + wave;      // tile = mt*12 + ng
  if (tile >= Mtiles * 12) return;
  const int ng = tile % 12, mt = tile / 12;
  const int n0 = ng * 64, m0 = mt * 16;
  int rowA = m0 + lo; if (rowA > M - 1) rowA = M - 1;
  const u16* arow = A + (size_t)rowA * 768;
  const u16* brow = WT + (size_t)(n0 + lo) * 768;   // f-th tile at +f*16*768

  v8f acc[4] = {v8f{}, v8f{}, v8f{}, v8f{}};
  Frag afr[2];
  Frag bfr[2][4];

  auto load_step = [&](int k0, int ph) {
    afr[ph].q[0] = *(const v4u*)(arow + k0 + hi * 16);
    afr[ph].q[1] = *(const v4u*)(arow + k0 + hi * 16 + 8);
    #pragma unroll
    for (int f = 0; f < 4; ++f)
      bfr[ph][f].o = *(const v8u*)(brow + (size_t)f * 12288 + k0 + hi * 16);
  };

  load_step(0, 0);
  #pragma unroll
  for (int k0 = 0; k0 < 768; k0 += 32) {
    const int ph = (k0 >> 5) & 1;
    if (k0 + 32 < 768) load_step(k0 + 32, ph ^ 1);
    if ((k0 & 96) == 0) __builtin_prefetch(arow + k0 + 256, 0, 1);
    acc[0] = wmma_bf(afr[ph], bfr[ph][0], acc[0]);
    acc[1] = wmma_bf(afr[ph], bfr[ph][1], acc[1]);
    acc[2] = wmma_bf(afr[ph], bfr[ph][2], acc[2]);
    acc[3] = wmma_bf(afr[ph], bfr[ph][3], acc[3]);
  }
  #pragma unroll
  for (int f = 0; f < 4; ++f) {
    const float bn = bias ? bias[n0 + f * 16 + lo] : 0.f;
    #pragma unroll
    for (int v = 0; v < 8; ++v) {
      int row = m0 + v + hi * 8;
      if (row < M) {
        float val = acc[f][v] + bn;
        size_t o = (size_t)row * 768 + n0 + f * 16 + lo;
        if (outF) outF[o] = val;
        if (outB) outB[o] = f2bf(val);
      }
    }
  }
}

// ----------------------- flash self-attention (WMMA) ------------------------
// grid (32 qtiles, 12 heads, 8 batches), 1 wave per block.
__global__ void __launch_bounds__(32)
flash_self_attn(const u16* __restrict__ Qb, const u16* __restrict__ Kb,
                const u16* __restrict__ Vt, const int* __restrict__ mask,
                u16* __restrict__ ys) {
  __shared__ __align__(16) u16 plds[16 * 32];
  const int lane = threadIdx.x;
  const int lo = lane & 15, hi = lane >> 4;
  const int qt = blockIdx.x, h = blockIdx.y, b = blockIdx.z;
  const float inv = 0.125f;

  int qi = qt * 16 + lo; if (qi > 510) qi = 510;
  const u16* qrow = Qb + ((size_t)(b * 512 + 1 + qi)) * 768 + h * 64;
  Frag a0, a1;
  a0.q[0] = *(const v4u*)(qrow + hi * 16);
  a0.q[1] = *(const v4u*)(qrow + hi * 16 + 8);
  a1.q[0] = *(const v4u*)(qrow + 32 + hi * 16);
  a1.q[1] = *(const v4u*)(qrow + 32 + hi * 16 + 8);

  v8f acc[4] = {v8f{}, v8f{}, v8f{}, v8f{}};
  float rm[8], rl[8];
  #pragma unroll
  for (int v = 0; v < 8; ++v) { rm[v] = -3.0e38f; rl[v] = 0.f; }

  for (int kb = 0; kb < 16; ++kb) {
    v8f s0 = {}, s1 = {};
    {
      int key = kb * 32 + lo;
      const u16* krow = Kb + ((size_t)(b * 512 + key)) * 768 + h * 64;
      Frag b0, b1;
      b0.o = *(const v8u*)(krow + hi * 16);
      b1.o = *(const v8u*)(krow + 32 + hi * 16);
      s0 = wmma_bf(a0, b0, s0);
      s0 = wmma_bf(a1, b1, s0);
    }
    {
      int key = kb * 32 + 16 + lo;
      const u16* krow = Kb + ((size_t)(b * 512 + key)) * 768 + h * 64;
      Frag b0, b1;
      b0.o = *(const v8u*)(krow + hi * 16);
      b1.o = *(const v8u*)(krow + 32 + hi * 16);
      s1 = wmma_bf(a0, b0, s1);
      s1 = wmma_bf(a1, b1, s1);
    }
    const int mk0 = mask[b * 512 + kb * 32 + lo];
    const int mk1 = mask[b * 512 + kb * 32 + 16 + lo];
    float p0[8], p1[8];
    #pragma unroll
    for (int v = 0; v < 8; ++v) {
      float x0 = mk0 ? s0[v] * inv : -1.0e9f;
      float x1 = mk1 ? s1[v] * inv : -1.0e9f;
      float mx = fmaxf(x0, x1);
      for (int d = 1; d < 16; d <<= 1) mx = fmaxf(mx, __shfl_xor(mx, d, 32));
      float nm = fmaxf(rm[v], mx);
      float corr = __expf(rm[v] - nm);
      float e0 = __expf(x0 - nm);
      float e1 = __expf(x1 - nm);
      float rs = e0 + e1;
      for (int d = 1; d < 16; d <<= 1) rs += __shfl_xor(rs, d, 32);
      rl[v] = rl[v] * corr + rs;
      rm[v] = nm;
      #pragma unroll
      for (int f = 0; f < 4; ++f) acc[f][v] *= corr;
      p0[v] = e0; p1[v] = e1;
    }
    __syncthreads();
    #pragma unroll
    for (int v = 0; v < 8; ++v) {
      plds[(v + hi * 8) * 32 + lo]      = f2bf(p0[v]);
      plds[(v + hi * 8) * 32 + 16 + lo] = f2bf(p1[v]);
    }
    __syncthreads();
    Frag ap;
    ap.q[0] = *(const v4u*)(plds + lo * 32 + hi * 16);
    ap.q[1] = *(const v4u*)(plds + lo * 32 + hi * 16 + 8);
    #pragma unroll
    for (int f = 0; f < 4; ++f) {
      const u16* vrow = Vt + ((size_t)(b * 768 + h * 64 + f * 16 + lo)) * 512
                          + kb * 32 + hi * 16;
      Frag bv; bv.o = *(const v8u*)vrow;
      acc[f] = wmma_bf(ap, bv, acc[f]);
    }
  }
  #pragma unroll
  for (int v = 0; v < 8; ++v) {
    int rq = qt * 16 + v + hi * 8;
    if (rq < 511) {
      float il = 1.0f / rl[v];
      size_t base = ((size_t)(b * 511 + rq)) * 768 + h * 64 + lo;
      ys[base + 0]  = f2bf(acc[0][v] * il);
      ys[base + 16] = f2bf(acc[1][v] * il);
      ys[base + 32] = f2bf(acc[2][v] * il);
      ys[base + 48] = f2bf(acc[3][v] * il);
    }
  }
}

// -------------------------- class-token attention ---------------------------
__global__ void __launch_bounds__(256)
class_attn(const u16* __restrict__ Qb, const u16* __restrict__ kcb,
           const u16* __restrict__ Vb, const int* __restrict__ mask,
           u16* __restrict__ yc) {
  __shared__ float lg[12 * 512];
  const int t = threadIdx.x, b = blockIdx.x;
  for (int idx = t; idx < 12 * 512; idx += 256) {
    int h = idx >> 9, key = idx & 511;
    const u16* q = Qb + ((size_t)(b * 512)) * 768 + h * 64;
    const u16* k = kcb + ((size_t)(b * 512 + key)) * 768 + h * 64;
    float s = 0.f;
    for (int d = 0; d < 64; ++d) s += bf2f(q[d]) * bf2f(k[d]);
    s *= 0.125f;
    if (mask[b * 512 + key] == 0) s = -1.0e9f;
    lg[idx] = s;
  }
  __syncthreads();
  if (t < 12) {
    float mx = -3.0e38f;
    for (int k = 0; k < 512; ++k) mx = fmaxf(mx, lg[t * 512 + k]);
    float sm = 0.f;
    for (int k = 0; k < 512; ++k) {
      float e = __expf(lg[t * 512 + k] - mx);
      lg[t * 512 + k] = e; sm += e;
    }
    float is = 1.f / sm;
    for (int k = 0; k < 512; ++k) lg[t * 512 + k] *= is;
  }
  __syncthreads();
  for (int o = t; o < 768; o += 256) {
    int h = o >> 6;
    float acc = 0.f;
    for (int k = 0; k < 512; ++k)
      acc += lg[h * 512 + k] * bf2f(Vb[((size_t)(b * 512 + k)) * 768 + o]);
    yc[b * 768 + o] = f2bf(acc);
  }
}

// --------------------------- neighbour attention ----------------------------
__global__ void __launch_bounds__(256)
neigh_attn(const u16* __restrict__ Qnb, const u16* __restrict__ knb,
           const u16* __restrict__ Vnb, u16* __restrict__ out_n) {
  __shared__ float qs[5 * 768], ks[5 * 768], vs[5 * 768];
  __shared__ float ps[12 * 5 * 5];
  const int t = threadIdx.x;
  const size_t base = (size_t)blockIdx.x * 5;
  for (int i = t; i < 5 * 768; i += 256) {
    size_t o = base * 768 + i;
    qs[i] = bf2f(Qnb[o]); ks[i] = bf2f(knb[o]); vs[i] = bf2f(Vnb[o]);
  }
  __syncthreads();
  for (int idx = t; idx < 60; idx += 256) {
    int h = idx / 5, i = idx % 5;
    const float* q = qs + i * 768 + h * 64;
    float l[5], mx = -3.0e38f;
    #pragma unroll
    for (int j = 0; j < 5; ++j) {
      const float* k = ks + j * 768 + h * 64;
      float s = 0.f;
      for (int d = 0; d < 64; ++d) s += q[d] * k[d];
      l[j] = s * 0.125f; mx = fmaxf(mx, l[j]);
    }
    float sm = 0.f;
    #pragma unroll
    for (int j = 0; j < 5; ++j) { l[j] = __expf(l[j] - mx); sm += l[j]; }
    float is = 1.f / sm;
    #pragma unroll
    for (int j = 0; j < 5; ++j) ps[(h * 5 + i) * 5 + j] = l[j] * is;
  }
  __syncthreads();
  for (int o = t; o < 5 * 768; o += 256) {
    int i = o / 768, col = o % 768, h = col >> 6;
    float acc = 0.f;
    #pragma unroll
    for (int j = 0; j < 5; ++j)
      acc += ps[(h * 5 + i) * 5 + j] * vs[j * 768 + col];
    out_n[base * 768 + o] = f2bf(acc);
  }
}

// -------------------------- LayerNorm combine -------------------------------
DEV float block_reduce_sum(float v, float* red) {
  int t = threadIdx.x;
  for (int d = 1; d < 32; d <<= 1) v += __shfl_xor(v, d, 32);
  if ((t & 31) == 0) red[t >> 5] = v;
  __syncthreads();
  float s = 0.f;
  if (t < 8) {
    s = red[t];
    for (int d = 1; d < 8; d <<= 1) s += __shfl_xor(s, d, 32);
  }
  if (t == 0) red[0] = s;
  __syncthreads();
  float r = red[0];
  __syncthreads();
  return r;
}

__global__ void __launch_bounds__(256)
combine_y(const float* __restrict__ x, const float* __restrict__ ycp,
          const float* __restrict__ ysp, const u16* __restrict__ out_p,
          const float* __restrict__ g, const float* __restrict__ bb,
          float* __restrict__ dout) {
  __shared__ float red[8];
  const int t = threadIdx.x, row = blockIdx.x;  // 0..4095
  const int b = row >> 9, tt = row & 511;
  float vals[3];
  #pragma unroll
  for (int j = 0; j < 3; ++j) {
    int c = t + j * 256;
    float pre;
    if (tt == 0) {
      pre = ycp[b * 768 + c];
    } else {
      size_t r = (size_t)(b * 511 + tt - 1);
      pre = 0.5f * bf2f(out_p[(r * 5) * 768 + c]) + 0.5f * ysp[r * 768 + c];
    }
    vals[j] = pre + x[(size_t)row * 768 + c];
  }
  float mean = block_reduce_sum(vals[0] + vals[1] + vals[2], red) * (1.f / 768.f);
  float vsum = 0.f;
  #pragma unroll
  for (int j = 0; j < 3; ++j) { float d = vals[j] - mean; vsum += d * d; }
  float var = block_reduce_sum(vsum, red) * (1.f / 768.f);
  float rstd = rsqrtf(var + 1e-12f);
  #pragma unroll
  for (int j = 0; j < 3; ++j) {
    int c = t + j * 256;
    dout[(size_t)row * 768 + c] = (vals[j] - mean) * rstd * g[c] + bb[c];
  }
}

__global__ void __launch_bounds__(256)
combine_neigh(const float* __restrict__ nrep, const u16* __restrict__ out_p,
              const float* __restrict__ g, const float* __restrict__ bb,
              float* __restrict__ dout) {
  __shared__ float red[8];
  const int t = threadIdx.x, r = blockIdx.x;   // 0..16351
  const int bt = r >> 2, jj = r & 3;
  const size_t oprow = ((size_t)bt * 5 + 1 + jj) * 768;
  float vals[3];
  #pragma unroll
  for (int j = 0; j < 3; ++j) {
    int c = t + j * 256;
    vals[j] = bf2f(out_p[oprow + c]) + nrep[(size_t)r * 768 + c];
  }
  float mean = block_reduce_sum(vals[0] + vals[1] + vals[2], red) * (1.f / 768.f);
  float vsum = 0.f;
  #pragma unroll
  for (int j = 0; j < 3; ++j) { float d = vals[j] - mean; vsum += d * d; }
  float var = block_reduce_sum(vsum, red) * (1.f / 768.f);
  float rstd = rsqrtf(var + 1e-12f);
  #pragma unroll
  for (int j = 0; j < 3; ++j) {
    int c = t + j * 256;
    dout[(size_t)r * 768 + c] = (vals[j] - mean) * rstd * g[c] + bb[c];
  }
}

// ------------------------------- host side ----------------------------------

static inline unsigned cdiv(size_t a, size_t b) { return (unsigned)((a + b - 1) / b); }

extern "C" void kernel_launch(void* const* d_in, const int* in_sizes, int n_in,
                              void* d_out, int out_size, void* d_ws, size_t ws_size,
                              hipStream_t stream) {
  (void)in_sizes; (void)n_in; (void)out_size; (void)ws_size;
  const float* x    = (const float*)d_in[0];
  const int*   mask = (const int*)  d_in[1];
  const float* nrep = (const float*)d_in[2];
  const float* Wq = (const float*)d_in[3];  const float* bq = (const float*)d_in[4];
  const float* Wk = (const float*)d_in[5];  const float* bk = (const float*)d_in[6];
  const float* Wv = (const float*)d_in[7];  const float* bv = (const float*)d_in[8];
  const float* Wp = (const float*)d_in[9];  const float* bp = (const float*)d_in[10];
  const float* Wd = (const float*)d_in[11]; const float* bd = (const float*)d_in[12];
  const float* Wu = (const float*)d_in[13]; const float* bu = (const float*)d_in[14];
  const float* lng = (const float*)d_in[15]; const float* lnb = (const float*)d_in[16];
  float* out = (float*)d_out;

  const int NX = 4096;            // B*T rows
  const int NC = 20440;           // B*(T-1)*5 candidate rows
  const size_t W2 = (size_t)768 * 768;

  char* base = (char*)d_ws; size_t off = 0;
  auto take = [&](size_t bytes) -> void* {
    void* r = base + off;
    off = (off + bytes + 255) & ~(size_t)255;
    return r;
  };
  u16* WqT  = (u16*)take(W2 * 2);
  u16* WkT  = (u16*)take(W2 * 2);
  u16* WvT  = (u16*)take(W2 * 2);
  u16* WpT  = (u16*)take(W2 * 2);
  u16* xb   = (u16*)take((size_t)NX * 768 * 2);
  u16* candb= (u16*)take((size_t)NC * 768 * 2);     // reused later as out_p
  u16* Qb   = (u16*)take((size_t)NX * 768 * 2);
  u16* Kb   = (u16*)take((size_t)NX * 768 * 2);
  u16* Vb   = (u16*)take((size_t)NX * 768 * 2);
  u16* kcb  = (u16*)take((size_t)NX * 768 * 2);
  u16* Vtb  = (u16*)take((size_t)8 * 768 * 512 * 2);
  u16* Qnb  = (u16*)take((size_t)NC * 768 * 2);
  u16* Vnb  = (u16*)take((size_t)NC * 768 * 2);
  u16* knb  = (u16*)take((size_t)NC * 768 * 2);
  float* xd    = (float*)take((size_t)NX * 8 * 4);
  float* candd = (float*)take((size_t)NC * 8 * 4);
  u16* ysat = (u16*)take((size_t)4088 * 768 * 2);
  u16* ycat = (u16*)take((size_t)8 * 768 * 2);
  u16* outn = (u16*)take((size_t)NC * 768 * 2);
  float* ysp = (float*)take((size_t)4088 * 768 * 4);
  float* ycp = (float*)take((size_t)8 * 768 * 4);
  u16* outp = candb;   // disjoint lifetime: candb dead before outp written

  auto gemm = [&](const u16* A, int M, const u16* WT, const float* bias,
                  float* oF, u16* oB) {
    int mt = (M + 15) >> 4;
    unsigned blocks = (unsigned)((mt * 12 + 7) / 8);
    gemm_bf16_768<<<blocks, 256, 0, stream>>>(A, M, WT, bias, oF, oB);
  };

  // 1) weight + activation conversion
  cvt_T768_kernel<<<cdiv(W2, 256), 256, 0, stream>>>(Wq, WqT);
  cvt_T768_kernel<<<cdiv(W2, 256), 256, 0, stream>>>(Wk, WkT);
  cvt_T768_kernel<<<cdiv(W2, 256), 256, 0, stream>>>(Wv, WvT);
  cvt_T768_kernel<<<cdiv(W2, 256), 256, 0, stream>>>(Wp, WpT);
  cvt_bf16_kernel<<<cdiv((size_t)NX * 768, 256), 256, 0, stream>>>(x, xb, NX * 768);
  gather_cand_kernel<<<cdiv((size_t)NC * 768, 256), 256, 0, stream>>>(x, nrep, candb);

  // 2) projections (WMMA)
  gemm(xb, NX, WqT, bq, nullptr, Qb);
  gemm(xb, NX, WkT, bk, nullptr, Kb);
  gemm(xb, NX, WvT, bv, nullptr, Vb);
  transpose_vt_kernel<<<cdiv((size_t)8 * 768 * 512, 256), 256, 0, stream>>>(Vb, Vtb);
  gemv_rank8_kernel<<<cdiv((size_t)NX * 8, 256), 256, 0, stream>>>(xb, NX, Wd, bd, xd);
  lowrank_up_kernel<<<cdiv((size_t)NX * 768, 256), 256, 0, stream>>>(xd, NX, Wu, bu, kcb);
  gemm(candb, NC, WqT, bq, nullptr, Qnb);
  gemm(candb, NC, WvT, bv, nullptr, Vnb);
  gemv_rank8_kernel<<<cdiv((size_t)NC * 8, 256), 256, 0, stream>>>(candb, NC, Wd, bd, candd);
  lowrank_up_kernel<<<cdiv((size_t)NC * 768, 256), 256, 0, stream>>>(candd, NC, Wu, bu, knb);

  // 3) attention
  flash_self_attn<<<dim3(32, 12, 8), 32, 0, stream>>>(Qb, Kb, Vtb, mask, ysat);
  class_attn<<<8, 256, 0, stream>>>(Qb, kcb, Vb, mask, ycat);
  neigh_attn<<<4088, 256, 0, stream>>>(Qnb, knb, Vnb, outn);

  // 4) output projections (WMMA)
  gemm(ysat, 4088, WpT, bp, ysp, nullptr);
  gemm(ycat, 8,    WpT, bp, ycp, nullptr);
  gemm(outn, NC,   WpT, bp, nullptr, outp);

  // 5) combine + LayerNorm
  combine_y<<<4096, 256, 0, stream>>>(x, ycp, ysp, outp, lng, lnb, out);
  combine_neigh<<<16352, 256, 0, stream>>>(nrep, outp, lng, lnb,
                                           out + (size_t)8 * 512 * 768);
}
